// EncoderBlock_2568390443463
// MI455X (gfx1250) — matrix operand order repaired and checked by reading, not compiled, not run on hardware
//
// QANet-style encoder block for MI455X (gfx1250, wave32, WMMA).
//
// Roofline: ~56 GFLOP across GEMMs/attention; activation traffic ~150MB ->
// ~6.5us HBM floor at 23.3 TB/s. All weights + staged activations live in the
// 192MB L2. v3 changes vs v2:
//  - __launch_bounds__(32, 1) on single-wave WMMA kernels: v2's 64x64 tile
//    (128 acc VGPRs) spilled to scratch inside the K-loop under the default
//    occupancy-driven VGPR budget; one-wave-per-EU budget fits ~230 VGPRs
//    with no spills, and 16 independent WMMA chains give enough ILP.

#include <hip/hip_runtime.h>

typedef __attribute__((ext_vector_type(16))) _Float16 v16h;
typedef __attribute__((ext_vector_type(8)))  _Float16 v8h;
typedef __attribute__((ext_vector_type(8)))  float    v8f;

#define DEV static __device__ __forceinline__

static constexpr int B_  = 8;
static constexpr int D_  = 512;
static constexpr int L_  = 1024;
static constexpr int N_  = B_ * L_;   // 8192 tokens
static constexpr int NH_ = 8;
static constexpr int DH_ = D_ / NH_;  // 64

DEV unsigned lane_id() { return threadIdx.x & 31u; }

DEV v8f wmma16x16x32(v16h a, v16h b, v8f c) {
  // (neg_a, A, neg_b, B, c_mod, C, reuse_a, reuse_b)
  return __builtin_amdgcn_wmma_f32_16x16x32_f16(false, a, false, b, (short)0, c,
                                                false, false);
}

// A fragment (16x32 MxK), element (r,k) at tile[r*ld + k].
// Lane r = lane&15, half hh = lane>>4: f[j] = A[r, hh*8+j], f[8+j] = A[r, hh*8+16+j].
DEV v16h load_fragA(const _Float16* tile, int ld) {
  unsigned l = lane_id();
  const _Float16* p = tile + (size_t)(l & 15u) * (size_t)ld + ((l >> 4) & 1u) * 8;
  v8h lo = *(const v8h*)(p);
  v8h hi = *(const v8h*)(p + 16);
  v16h f;
#pragma unroll
  for (int j = 0; j < 8; ++j) { f[j] = lo[j]; f[j + 8] = hi[j]; }
  return f;
}

// B fragment (32x16 KxN), element (k,n) at tile[n*ld + k] (K-contiguous).
// Lane n = lane&15, half hh = lane>>4: f[j] = B[hh*8 + j, n], f[8+j] = B[hh*8+16+j, n].
DEV v16h load_fragB(const _Float16* tile, int ld) {
  unsigned l = lane_id();
  const _Float16* p = tile + (size_t)(l & 15u) * (size_t)ld + ((l >> 4) & 1u) * 8;
  v8h lo = *(const v8h*)(p);
  v8h hi = *(const v8h*)(p + 16);
  v16h f;
#pragma unroll
  for (int j = 0; j < 8; ++j) { f[j] = lo[j]; f[j + 8] = hi[j]; }
  return f;
}

// 64x64 C tile per wave, K = 512. A: weights [O,512] row-major at row o0.
// Bm: activations [N,512] at token n0. acc[mi][ni] = rows o0+16mi, cols n0+16ni.
DEV void gemm_core_64x64(const _Float16* __restrict__ A,
                         const _Float16* __restrict__ Bm, v8f acc[4][4]) {
#pragma unroll 4
  for (int k = 0; k < 512; k += 32) {
    v16h a[4], bf[4];
#pragma unroll
    for (int i = 0; i < 4; ++i) a[i]  = load_fragA(A  + (size_t)(16 * i) * 512 + k, 512);
#pragma unroll
    for (int i = 0; i < 4; ++i) bf[i] = load_fragB(Bm + (size_t)(16 * i) * 512 + k, 512);
#pragma unroll
    for (int mi = 0; mi < 4; ++mi)
#pragma unroll
      for (int ni = 0; ni < 4; ++ni)
        acc[mi][ni] = wmma16x16x32(a[mi], bf[ni], acc[mi][ni]);
  }
}

// ---------------------------------------------------------------- utilities

__global__ void k_cvt_f32_f16(const float* __restrict__ s, _Float16* __restrict__ d,
                              int n) {
  int i = blockIdx.x * 256 + threadIdx.x;
  if (i < n) d[i] = (_Float16)s[i];
}

// out[b,d,l] = x[b,d,l] + pos(d,l)
__global__ void k_pos(const float* __restrict__ x, float* __restrict__ out) {
  int idx = blockIdx.x * 256 + threadIdx.x;  // over B*D*L = 4M
  int l = idx & (L_ - 1);
  int d = (idx >> 10) & (D_ - 1);
  const float log_inc = 0.03612013f;  // ln(10000)/255
  float t;
  if (d < 256) t = __sinf((float)l * __expf(-(float)d * log_inc));
  else         t = __cosf((float)l * __expf(-(float)(d - 256) * log_inc));
  out[idx] = x[idx] + t;
}

// LayerNorm over channel dim of [B,D,L] f32 -> [N,D] f16 (token-major).
// Block = (32 tokens) x (8 channel groups of 64); LDS tree for mean/var.
__global__ void k_ln(const float* __restrict__ in, const float* __restrict__ g,
                     const float* __restrict__ bta, _Float16* __restrict__ out) {
  __shared__ float s_sum[8][32];
  __shared__ float s_ssum[8][32];
  __shared__ float s_mean[32];
  __shared__ float s_inv[32];
  int tx = threadIdx.x, ty = threadIdx.y;
  int n = blockIdx.x * 32 + tx;  // token id
  int b = n >> 10, l = n & (L_ - 1);
  const float* p = in + (size_t)b * D_ * L_ + l;
  float s = 0.f, ss = 0.f;
  int d0 = ty * 64;
  for (int d = d0; d < d0 + 64; ++d) {
    float v = p[(size_t)d * L_];
    s += v; ss += v * v;
  }
  s_sum[ty][tx] = s; s_ssum[ty][tx] = ss;
  __syncthreads();
  if (ty == 0) {
    float S = 0.f, SS = 0.f;
#pragma unroll
    for (int j = 0; j < 8; ++j) { S += s_sum[j][tx]; SS += s_ssum[j][tx]; }
    float mean = S * (1.0f / D_);
    float var  = SS * (1.0f / D_) - mean * mean;
    s_mean[tx] = mean;
    s_inv[tx]  = rsqrtf(var + 1e-5f);
  }
  __syncthreads();
  float mean = s_mean[tx], inv = s_inv[tx];
  _Float16* q = out + (size_t)n * D_;
  for (int d = d0; d < d0 + 64; ++d) {
    float v = p[(size_t)d * L_];
    q[d] = (_Float16)(g[d] * (v - mean) * inv + bta[d]);
  }
}

// Depthwise conv k=7 along l on [N,D] f16 (zero-padded), f32 accumulate.
__global__ void k_dw(const _Float16* __restrict__ ln, const float* __restrict__ w,
                     _Float16* __restrict__ X) {
  size_t idx = (size_t)blockIdx.x * 256 + threadIdx.x;  // over N*D
  int d = (int)(idx & (D_ - 1));
  int n = (int)(idx >> 9);
  int b = n >> 10, l = n & (L_ - 1);
  float acc = 0.f;
#pragma unroll
  for (int t = 0; t < 7; ++t) {
    int lt = l + t - 3;
    if (lt >= 0 && lt < L_)
      acc += (float)ln[((size_t)(b * L_ + lt)) * D_ + d] * w[d * 7 + t];
  }
  X[idx] = (_Float16)acc;
}

// ------------------------------------------------------------ GEMM kernels

// out[b,o,l] = relu(W@X + bias) + out[b,o,l]  (in-place residual)
__global__ __launch_bounds__(32, 1)
void k_pw_gemm(const _Float16* __restrict__ W, const _Float16* __restrict__ X,
               const float* __restrict__ bias, float* out) {
  int n0 = blockIdx.x * 64, o0 = blockIdx.y * 64;
  v8f acc[4][4] = {};
  gemm_core_64x64(W + (size_t)o0 * 512, X + (size_t)n0 * 512, acc);
  unsigned l = lane_id(); int col = l & 15, hh = (int)(l >> 4);
#pragma unroll
  for (int ni = 0; ni < 4; ++ni) {
    int n = n0 + ni * 16 + col;
    int b = n >> 10, ll = n & (L_ - 1);
#pragma unroll
    for (int mi = 0; mi < 4; ++mi) {
#pragma unroll
      for (int v = 0; v < 8; ++v) {
        int o = o0 + mi * 16 + v + 8 * hh;
        float g = fmaxf(acc[mi][ni][v] + bias[o], 0.0f);
        size_t ad = ((size_t)(b * D_ + o)) * L_ + ll;
        out[ad] = g + out[ad];
      }
    }
  }
}

// H1[n,o] = relu(W@X + bias) as f16 [N,D]
__global__ __launch_bounds__(32, 1)
void k_ffn1_gemm(const _Float16* __restrict__ W, const _Float16* __restrict__ X,
                 const float* __restrict__ bias, _Float16* __restrict__ H) {
  int n0 = blockIdx.x * 64, o0 = blockIdx.y * 64;
  v8f acc[4][4] = {};
  gemm_core_64x64(W + (size_t)o0 * 512, X + (size_t)n0 * 512, acc);
  unsigned l = lane_id(); int col = l & 15, hh = (int)(l >> 4);
#pragma unroll
  for (int ni = 0; ni < 4; ++ni) {
    int n = n0 + ni * 16 + col;
#pragma unroll
    for (int mi = 0; mi < 4; ++mi) {
#pragma unroll
      for (int v = 0; v < 8; ++v) {
        int o = o0 + mi * 16 + v + 8 * hh;
        H[(size_t)n * D_ + o] = (_Float16)fmaxf(acc[mi][ni][v] + bias[o], 0.0f);
      }
    }
  }
}

// dout[b,o,l] = W@X + bias + res[b,o,l]
__global__ __launch_bounds__(32, 1)
void k_ffn2_gemm(const _Float16* __restrict__ W, const _Float16* __restrict__ X,
                 const float* __restrict__ bias, const float* __restrict__ res,
                 float* __restrict__ dout) {
  int n0 = blockIdx.x * 64, o0 = blockIdx.y * 64;
  v8f acc[4][4] = {};
  gemm_core_64x64(W + (size_t)o0 * 512, X + (size_t)n0 * 512, acc);
  unsigned l = lane_id(); int col = l & 15, hh = (int)(l >> 4);
#pragma unroll
  for (int ni = 0; ni < 4; ++ni) {
    int n = n0 + ni * 16 + col;
    int b = n >> 10, ll = n & (L_ - 1);
#pragma unroll
    for (int mi = 0; mi < 4; ++mi) {
#pragma unroll
      for (int v = 0; v < 8; ++v) {
        int o = o0 + mi * 16 + v + 8 * hh;
        size_t ad = ((size_t)(b * D_ + o)) * L_ + ll;
        dout[ad] = acc[mi][ni][v] + bias[o] + res[ad];
      }
    }
  }
}

// query projection: Q[b,h,l,dh] = 0.125 * (Wq @ LN)   (pre-scaled by dh^-0.5)
__global__ __launch_bounds__(32, 1)
void k_q_gemm(const _Float16* __restrict__ W, const _Float16* __restrict__ X,
              _Float16* __restrict__ Q) {
  int n0 = blockIdx.x * 64, o0 = blockIdx.y * 64;
  v8f acc[4][4] = {};
  gemm_core_64x64(W + (size_t)o0 * 512, X + (size_t)n0 * 512, acc);
  unsigned l = lane_id(); int col = l & 15, hh = (int)(l >> 4);
#pragma unroll
  for (int ni = 0; ni < 4; ++ni) {
    int n = n0 + ni * 16 + col;
    int b = n >> 10, ll = n & (L_ - 1);
#pragma unroll
    for (int mi = 0; mi < 4; ++mi) {
#pragma unroll
      for (int v = 0; v < 8; ++v) {
        int o = o0 + mi * 16 + v + 8 * hh;
        int h = o >> 6, dh = o & 63;
        Q[(((size_t)(b * NH_ + h)) * L_ + ll) * DH_ + dh] =
            (_Float16)(acc[mi][ni][v] * 0.125f);
      }
    }
  }
}

// memory projection (2D x D): rows [0,512) -> K[b,h,l,dh]; rows [512,1024) -> Vt[b,h,dh,l]
__global__ __launch_bounds__(32, 1)
void k_mem_gemm(const _Float16* __restrict__ W, const _Float16* __restrict__ X,
                _Float16* __restrict__ K, _Float16* __restrict__ Vt) {
  int n0 = blockIdx.x * 64, o0 = blockIdx.y * 64;
  v8f acc[4][4] = {};
  gemm_core_64x64(W + (size_t)o0 * 512, X + (size_t)n0 * 512, acc);
  unsigned l = lane_id(); int col = l & 15, hh = (int)(l >> 4);
#pragma unroll
  for (int ni = 0; ni < 4; ++ni) {
    int n = n0 + ni * 16 + col;
    int b = n >> 10, ll = n & (L_ - 1);
#pragma unroll
    for (int mi = 0; mi < 4; ++mi) {
#pragma unroll
      for (int v = 0; v < 8; ++v) {
        int o = o0 + mi * 16 + v + 8 * hh;
        _Float16 val = (_Float16)acc[mi][ni][v];
        if (o < D_) {
          int h = o >> 6, dh = o & 63;
          K[(((size_t)(b * NH_ + h)) * L_ + ll) * DH_ + dh] = val;
        } else {
          int ov = o - D_;
          int h = ov >> 6, dh = ov & 63;
          Vt[(((size_t)(b * NH_ + h)) * DH_ + dh) * L_ + ll] = val;
        }
      }
    }
  }
}

// ------------------------------------------------------------- attention
// One wave per (b, h, 32-query tile). Flash-style online softmax.
// Q,K: [B,H,L,64] f16; Vt: [B,H,64,L] f16; mask: [B,L] int; out += att (in place).
__global__ __launch_bounds__(32, 1)
void k_attn(const _Float16* __restrict__ Qb, const _Float16* __restrict__ Kb,
            const _Float16* __restrict__ Vtb, const int* __restrict__ mask,
            float* out) {
  int q0 = blockIdx.x * 32;
  int h  = blockIdx.y;
  int b  = blockIdx.z;
  unsigned l = lane_id(); int col = l & 15, hh = (int)(l >> 4);

  const _Float16* Qh = Qb  + (size_t)(b * NH_ + h) * L_ * DH_;
  const _Float16* Kh = Kb  + (size_t)(b * NH_ + h) * L_ * DH_;
  const _Float16* Vh = Vtb + (size_t)(b * NH_ + h) * DH_ * L_;
  const int* mb = mask + (size_t)b * L_;

  v16h qa[2][2];  // [row-tile][k-half]
#pragma unroll
  for (int r = 0; r < 2; ++r) {
    qa[r][0] = load_fragA(Qh + (size_t)(q0 + 16 * r) * DH_, DH_);
    qa[r][1] = load_fragA(Qh + (size_t)(q0 + 16 * r) * DH_ + 32, DH_);
  }

  v8f oacc[2][4];
#pragma unroll
  for (int r = 0; r < 2; ++r)
#pragma unroll
    for (int dt = 0; dt < 4; ++dt) oacc[r][dt] = {};
  float mrow[2][8], lrow[2][8];
#pragma unroll
  for (int r = 0; r < 2; ++r)
#pragma unroll
    for (int v = 0; v < 8; ++v) { mrow[r][v] = -1e30f; lrow[r][v] = 0.f; }

  __shared__ __align__(16) _Float16 P[32 * 32];

  for (int kb = 0; kb < L_; kb += 32) {
    // K fragments for 32 keys, K-dim = dh = 64 (two halves)
    v16h kb00 = load_fragB(Kh + (size_t)kb * DH_, DH_);
    v16h kb01 = load_fragB(Kh + (size_t)kb * DH_ + 32, DH_);
    v16h kb10 = load_fragB(Kh + (size_t)(kb + 16) * DH_, DH_);
    v16h kb11 = load_fragB(Kh + (size_t)(kb + 16) * DH_ + 32, DH_);

    float mk0 = mb[kb + col]      ? 0.f : -1e30f;
    float mk1 = mb[kb + 16 + col] ? 0.f : -1e30f;

#pragma unroll
    for (int r = 0; r < 2; ++r) {
      v8f s0 = {}, s1 = {};
      s0 = wmma16x16x32(qa[r][0], kb00, s0);
      s0 = wmma16x16x32(qa[r][1], kb01, s0);
      s1 = wmma16x16x32(qa[r][0], kb10, s1);
      s1 = wmma16x16x32(qa[r][1], kb11, s1);

#pragma unroll
      for (int v = 0; v < 8; ++v) {
        float a = s0[v] + mk0;
        float c = s1[v] + mk1;
        float mx = fmaxf(a, c);
#pragma unroll
        for (int off = 1; off < 16; off <<= 1)
          mx = fmaxf(mx, __shfl_xor(mx, off, 16));
        float mnew = fmaxf(mrow[r][v], mx);
        float sc   = __expf(mrow[r][v] - mnew);
        float p0   = __expf(a - mnew);
        float p1   = __expf(c - mnew);
        float rs   = p0 + p1;
#pragma unroll
        for (int off = 1; off < 16; off <<= 1)
          rs += __shfl_xor(rs, off, 16);
        lrow[r][v] = lrow[r][v] * sc + rs;
        mrow[r][v] = mnew;
        oacc[r][0][v] *= sc; oacc[r][1][v] *= sc;
        oacc[r][2][v] *= sc; oacc[r][3][v] *= sc;
        int row = r * 16 + v + 8 * hh;
        P[row * 32 + col]      = (_Float16)p0;
        P[row * 32 + 16 + col] = (_Float16)p1;
      }
    }
    // Same-wave DS ops are architecturally in-order; the wait + memory clobber
    // pins compiler ordering of the P stores before the A-frag loads.
    asm volatile("s_wait_dscnt 0x0" ::: "memory");

    v16h pa0 = load_fragA(P, 32);            // q-rows q0..q0+15  x 32 keys
    v16h pa1 = load_fragA(P + 16 * 32, 32);  // q-rows q0+16..+31 x 32 keys
#pragma unroll
    for (int dt = 0; dt < 4; ++dt) {
      v16h vb = load_fragB(Vh + (size_t)(16 * dt) * L_ + kb, L_);
      oacc[0][dt] = wmma16x16x32(pa0, vb, oacc[0][dt]);
      oacc[1][dt] = wmma16x16x32(pa1, vb, oacc[1][dt]);
    }
  }

#pragma unroll
  for (int r = 0; r < 2; ++r)
#pragma unroll
    for (int v = 0; v < 8; ++v) {
      float inv = 1.0f / lrow[r][v];
      int lq = q0 + r * 16 + v + 8 * hh;
#pragma unroll
      for (int dt = 0; dt < 4; ++dt) {
        int dm = h * DH_ + dt * 16 + col;
        size_t ad = ((size_t)(b * D_ + dm)) * L_ + lq;
        out[ad] = oacc[r][dt][v] * inv + out[ad];
      }
    }
}

// -------------------------------------------------------------- launcher

extern "C" void kernel_launch(void* const* d_in, const int* in_sizes, int n_in,
                              void* d_out, int out_size, void* d_ws, size_t ws_size,
                              hipStream_t stream) {
  (void)in_sizes; (void)n_in; (void)out_size; (void)ws_size;
  const float* x        = (const float*)d_in[0];
  const int*   mask     = (const int*)d_in[1];
  const float* dw_w     = (const float*)d_in[4];
  const float* pw_w     = (const float*)d_in[5];
  const float* pw_b     = (const float*)d_in[6];
  const float* normC_g  = (const float*)d_in[7];
  const float* normC_b  = (const float*)d_in[8];
  const float* mem_w    = (const float*)d_in[9];
  const float* query_w  = (const float*)d_in[10];
  const float* norm1_g  = (const float*)d_in[11];
  const float* norm1_b  = (const float*)d_in[12];
  const float* norm2_g  = (const float*)d_in[13];
  const float* norm2_b  = (const float*)d_in[14];
  const float* ffn1_w   = (const float*)d_in[15];
  const float* ffn1_b   = (const float*)d_in[16];
  const float* ffn2_w   = (const float*)d_in[17];
  const float* ffn2_b   = (const float*)d_in[18];
  float* dout = (float*)d_out;

  // workspace layout
  char* ws = (char*)d_ws;
  const size_t MB = 1024ull * 1024ull;
  float*    buf0 = (float*)(ws);                 // [B,D,L] f32 running activation (16MB)
  _Float16* lnb  = (_Float16*)(ws + 16 * MB);    // [N,D] f16 (8MB)
  _Float16* Xb   = (_Float16*)(ws + 24 * MB);    // [N,D] f16 (8MB)
  _Float16* Qb   = (_Float16*)(ws + 32 * MB);    // [B,H,L,64] f16 (8MB)
  _Float16* Kb   = (_Float16*)(ws + 40 * MB);    // [B,H,L,64] f16 (8MB)
  _Float16* Vtb  = (_Float16*)(ws + 48 * MB);    // [B,H,64,L] f16 (8MB)
  _Float16* Wh   = (_Float16*)(ws + 56 * MB);    // f16 weights (~4.5MB)
  _Float16* WhPW  = Wh;                          // 4 * 512*512
  _Float16* WhMem = Wh + 4 * D_ * D_;            // 1024*512
  _Float16* WhQ   = WhMem + 2 * D_ * D_;         // 512*512
  _Float16* WhF1  = WhQ + D_ * D_;
  _Float16* WhF2  = WhF1 + D_ * D_;

  // 0) convert weights to f16
  k_cvt_f32_f16<<<(4 * D_ * D_ + 255) / 256, 256, 0, stream>>>(pw_w, WhPW, 4 * D_ * D_);
  k_cvt_f32_f16<<<(2 * D_ * D_ + 255) / 256, 256, 0, stream>>>(mem_w, WhMem, 2 * D_ * D_);
  k_cvt_f32_f16<<<(D_ * D_ + 255) / 256, 256, 0, stream>>>(query_w, WhQ, D_ * D_);
  k_cvt_f32_f16<<<(D_ * D_ + 255) / 256, 256, 0, stream>>>(ffn1_w, WhF1, D_ * D_);
  k_cvt_f32_f16<<<(D_ * D_ + 255) / 256, 256, 0, stream>>>(ffn2_w, WhF2, D_ * D_);

  // 1) positional encoding
  k_pos<<<(B_ * D_ * L_) / 256, 256, 0, stream>>>(x, buf0);

  // 2) conv blocks: LN -> depthwise -> pointwise WMMA GEMM + relu + residual
  for (int i = 0; i < 4; ++i) {
    k_ln<<<N_ / 32, dim3(32, 8), 0, stream>>>(buf0, normC_g + i * D_, normC_b + i * D_, lnb);
    k_dw<<<(N_ * D_) / 256, 256, 0, stream>>>(lnb, dw_w + i * D_ * 7, Xb);
    k_pw_gemm<<<dim3(N_ / 64, D_ / 64), 32, 0, stream>>>(WhPW + (size_t)i * D_ * D_,
                                                         Xb, pw_b + i * D_, buf0);
  }

  // 3) attention: LN -> KV/Q projections -> flash attention (+ residual in place)
  k_ln<<<N_ / 32, dim3(32, 8), 0, stream>>>(buf0, norm1_g, norm1_b, lnb);
  k_mem_gemm<<<dim3(N_ / 64, (2 * D_) / 64), 32, 0, stream>>>(WhMem, lnb, Kb, Vtb);
  k_q_gemm<<<dim3(N_ / 64, D_ / 64), 32, 0, stream>>>(WhQ, lnb, Qb);
  k_attn<<<dim3(L_ / 32, NH_, B_), 32, 0, stream>>>(Qb, Kb, Vtb, mask, buf0);

  // 4) FFN: LN -> GEMM+relu -> GEMM + bias + residual -> d_out
  k_ln<<<N_ / 32, dim3(32, 8), 0, stream>>>(buf0, norm2_g, norm2_b, lnb);
  k_ffn1_gemm<<<dim3(N_ / 64, D_ / 64), 32, 0, stream>>>(WhF1, lnb, ffn1_b, Xb);
  k_ffn2_gemm<<<dim3(N_ / 64, D_ / 64), 32, 0, stream>>>(WhF2, Xb, ffn2_b, buf0, dout);
}
